// FixedKernelMultiAttention_14139032338897
// MI455X (gfx1250) — matrix operand order repaired and checked by reading, not compiled, not run on hardware
//
#include <hip/hip_runtime.h>
#include <hip/hip_bf16.h>
#include <math.h>

// ---------------- problem constants ----------------
static constexpr int kB = 4, kR = 32, kC = 512, kE = 512, kH = 8, kD = 64;
static constexpr int kMTOT = kB * kR * kC;   // 65536 rows of x flattened
static constexpr int kG    = kB * kR * kH;   // 1024 (b,r,h) groups

typedef _Float16 f16;
typedef __attribute__((ext_vector_type(16))) _Float16 v16h;
typedef __attribute__((ext_vector_type(8)))  _Float16 v8h;
typedef __attribute__((ext_vector_type(8)))  float    v8f;
typedef __attribute__((ext_vector_type(4)))  float    v4f;

// ---------------- WMMA helpers ----------------
// D = A(16x32 f16) x B(32x16 f16) + C(16x16 f32), wave32.
static __device__ __forceinline__ v8f wmma_f16(v16h a, v16h b, v8f c) {
    return __builtin_amdgcn_wmma_f32_16x16x32_f16(
        /*neg_a=*/false, a, /*neg_b=*/false, b,
        /*c_mod=*/(short)0, c, /*reuse_a=*/false, /*reuse_b=*/false);
}

// A fragment: lane l holds row m = l%16; half = l/16.
//   elements 0..7  -> K = half*8 + 0..7        (contiguous 16B)
//   elements 8..15 -> K = 16 + half*8 + 0..7   (contiguous 16B)
static __device__ __forceinline__ v16h load_frag_a(const f16* __restrict__ rowp,
                                                   int k0, int half) {
    v8h lo = *(const v8h*)(rowp + k0 + half * 8);
    v8h hi = *(const v8h*)(rowp + k0 + 16 + half * 8);
    return __builtin_shufflevector(lo, hi, 0,1,2,3,4,5,6,7,8,9,10,11,12,13,14,15);
}

// B fragment: lane l holds column n = l%16 (rowp = row n of B^T); half = l/16.
//   elements 0..7  -> K = half*8 + 0..7        (contiguous 16B)
//   elements 8..15 -> K = 16 + half*8 + 0..7   (contiguous 16B)
static __device__ __forceinline__ v16h load_frag_b(const f16* __restrict__ rowp,
                                                   int k0, int half) {
    v8h lo = *(const v8h*)(rowp + k0 + half * 8);
    v8h hi = *(const v8h*)(rowp + k0 + 16 + half * 8);
    return __builtin_shufflevector(lo, hi, 0,1,2,3,4,5,6,7,8,9,10,11,12,13,14,15);
}

// ---------------- f32 -> f16 convert (8 elems/thread, b128 in/out) ----------------
__global__ void k_f32_to_f16(const float* __restrict__ src, f16* __restrict__ dst, int n8) {
    int i = blockIdx.x * blockDim.x + threadIdx.x;
    if (i >= n8) return;
    const v4f a = *(const v4f*)(src + (size_t)i * 8);
    const v4f b = *(const v4f*)(src + (size_t)i * 8 + 4);
    v8h o;
#pragma unroll
    for (int j = 0; j < 4; ++j) { o[j] = (f16)a[j]; o[j + 4] = (f16)b[j]; }
    *(v8h*)(dst + (size_t)i * 8) = o;
}

// ---------------- fused QKV projection ----------------
// Y[m,n] = sum_k X[m,k] * W[n,k] + bias[n]; for q,k apply elu(y)+1.
// q stored [m, E]; k,v stored transposed: [(br*H+h)*D + d][c].
// grid: (M/128, E/128, 3), block 256 (8 waves: 4 along M (32 rows each), 2 along N).
__global__ void k_qkv(const f16* __restrict__ x16,
                      const f16* __restrict__ wq, const f16* __restrict__ wk,
                      const f16* __restrict__ wv,
                      const float* __restrict__ bq, const float* __restrict__ bk,
                      const float* __restrict__ bv,
                      f16* __restrict__ qd, f16* __restrict__ kd, f16* __restrict__ vd) {
    const int mode = blockIdx.z;
    const f16* w; const float* bias; f16* dst;
    if (mode == 0)      { w = wq; bias = bq; dst = qd; }
    else if (mode == 1) { w = wk; bias = bk; dst = kd; }
    else                { w = wv; bias = bv; dst = vd; }

    const int lane = threadIdx.x & 31;
    const int wave = threadIdx.x >> 5;
    const int nl = lane & 15;
    const int hf = lane >> 4;

    const int m0 = blockIdx.x * 128 + (wave & 3) * 32;   // 32 M-rows per wave
    const int n0 = blockIdx.y * 128 + (wave >> 2) * 64;  // 64 N-cols per wave

    const f16* rowA0 = x16 + (size_t)(m0 + nl) * kE;
    const f16* rowA1 = rowA0 + (size_t)16 * kE;
    const f16* rowB[4];
#pragma unroll
    for (int s = 0; s < 4; ++s)
        rowB[s] = w + (size_t)(n0 + s * 16 + nl) * kE;

    const v8f zero8 = {0.f,0.f,0.f,0.f,0.f,0.f,0.f,0.f};
    v8f acc0[4] = {zero8, zero8, zero8, zero8};
    v8f acc1[4] = {zero8, zero8, zero8, zero8};

    for (int k0 = 0; k0 < kE; k0 += 32) {
        if (k0 + 64 <= kE) {
            __builtin_prefetch(rowA0 + k0 + 32, 0, 3);
            __builtin_prefetch(rowA1 + k0 + 32, 0, 3);
        }
        v16h a0 = load_frag_a(rowA0, k0, hf);
        v16h a1 = load_frag_a(rowA1, k0, hf);
#pragma unroll
        for (int s = 0; s < 4; ++s) {
            v16h b = load_frag_b(rowB[s], k0, hf);
            acc0[s] = wmma_f16(a0, b, acc0[s]);
            acc1[s] = wmma_f16(a1, b, acc1[s]);
        }
    }

#pragma unroll
    for (int s = 0; s < 4; ++s) {
        const int n = n0 + s * 16 + nl;
        const float bval = bias[n];
#pragma unroll
        for (int t = 0; t < 2; ++t) {
#pragma unroll
            for (int j = 0; j < 8; ++j) {
                const int m = m0 + t * 16 + j + 8 * hf;
                float y = (t == 0 ? acc0[s][j] : acc1[s][j]) + bval;
                if (mode < 2) y = (y > 0.f) ? (y + 1.f) : __expf(y);  // elu(y)+1
                if (mode == 0) {
                    dst[(size_t)m * kE + n] = (f16)y;
                } else {
                    const int br = m >> 9, c = m & 511;   // m = br*C + c
                    const int hh = n >> 6, d = n & 63;    // n = hh*D + d
                    dst[((size_t)((((br << 3) | hh) << 6) | d) << 9) + c] = (f16)y;
                }
            }
        }
    }
}

// ---------------- ksum[g*D+d] = sum_c kT[(g*D+d)*C + c] ----------------
__global__ void k_ksum(const f16* __restrict__ kT, float* __restrict__ ksum) {
    int i = blockIdx.x * blockDim.x + threadIdx.x;   // 0 .. G*D-1
    const f16* row = kT + (size_t)i * kC;
    float s = 0.f;
    for (int c = 0; c < kC; c += 8) {
        v8h v = *(const v8h*)(row + c);
#pragma unroll
        for (int j = 0; j < 8; ++j) s += (float)v[j];
    }
    ksum[i] = s;
}

// ---------------- z[g*C+c] = 1/(q[c,:].ksum[g,:] + eps) ----------------
__global__ void k_z(const f16* __restrict__ q16, const float* __restrict__ ksum,
                    float* __restrict__ zb) {
    int i = blockIdx.x * blockDim.x + threadIdx.x;   // 0 .. G*C-1
    const int g = i >> 9, c = i & 511;
    const int br = g >> 3, hh = g & 7;
    const f16* qrow = q16 + (size_t)(br * kC + c) * kE + hh * kD;
    const float* ks = ksum + g * kD;
    float s = 0.f;
#pragma unroll 8
    for (int d = 0; d < kD; ++d) s += (float)qrow[d] * ks[d];
    zb[i] = 1.f / (s + 1e-6f);
}

// ---------------- per-group ktvT[e][d] = sum_c v[c,e]*k[c,d] ----------------
// A = vT (row e, K=c contiguous), B^T = kT (row d, K=c contiguous).
// grid: G, block 128 (4 waves along e).
__global__ void k_ktv(const f16* __restrict__ vT, const f16* __restrict__ kT,
                      f16* __restrict__ ktvT) {
    const int g = blockIdx.x;
    const int lane = threadIdx.x & 31;
    const int wave = threadIdx.x >> 5;
    const int nl = lane & 15;
    const int hf = lane >> 4;
    const int e0 = wave * 16;

    const f16* rowA = vT + (size_t)(g * kD + e0 + nl) * kC;
    const f16* rowB[4];
#pragma unroll
    for (int s = 0; s < 4; ++s)
        rowB[s] = kT + (size_t)(g * kD + s * 16 + nl) * kC;

    const v8f zero8 = {0.f,0.f,0.f,0.f,0.f,0.f,0.f,0.f};
    v8f acc[4] = {zero8, zero8, zero8, zero8};

    for (int k0 = 0; k0 < kC; k0 += 32) {
        if (k0 + 64 <= kC) __builtin_prefetch(rowA + k0 + 32, 0, 3);
        v16h a = load_frag_a(rowA, k0, hf);
#pragma unroll
        for (int s = 0; s < 4; ++s)
            acc[s] = wmma_f16(a, load_frag_b(rowB[s], k0, hf), acc[s]);
    }

#pragma unroll
    for (int s = 0; s < 4; ++s) {
        const int d = s * 16 + nl;
#pragma unroll
        for (int j = 0; j < 8; ++j) {
            const int e = e0 + j + 8 * hf;
            ktvT[(size_t)g * (kD * kD) + e * kD + d] = (f16)acc[s][j];
        }
    }
}

// ---------------- out_h[c,e] = z[c] * sum_d q[c,d]*ktv[d,e] ----------------
// A = q16 (row c, K=d contiguous at col offset hh*D); B^T = ktvT (row e, K=d contiguous).
// grid: (C/128, G), block 256 (8 waves along c).
__global__ void k_outh(const f16* __restrict__ q16, const f16* __restrict__ ktvT,
                       const float* __restrict__ zb, f16* __restrict__ outh) {
    const int g = blockIdx.y;
    const int br = g >> 3, hh = g & 7;
    const int lane = threadIdx.x & 31;
    const int wave = threadIdx.x >> 5;
    const int nl = lane & 15;
    const int hf = lane >> 4;
    const int c0 = blockIdx.x * 128 + wave * 16;

    const f16* rowA = q16 + (size_t)(br * kC + c0 + nl) * kE + hh * kD;
    const f16* rowB[4];
#pragma unroll
    for (int s = 0; s < 4; ++s)
        rowB[s] = ktvT + (size_t)g * (kD * kD) + (s * 16 + nl) * kD;

    const v8f zero8 = {0.f,0.f,0.f,0.f,0.f,0.f,0.f,0.f};
    v8f acc[4] = {zero8, zero8, zero8, zero8};

#pragma unroll
    for (int k0 = 0; k0 < kD; k0 += 32) {
        v16h a = load_frag_a(rowA, k0, hf);
#pragma unroll
        for (int s = 0; s < 4; ++s)
            acc[s] = wmma_f16(a, load_frag_b(rowB[s], k0, hf), acc[s]);
    }

#pragma unroll
    for (int s = 0; s < 4; ++s) {
        const int e = s * 16 + nl;
#pragma unroll
        for (int j = 0; j < 8; ++j) {
            const int c = c0 + j + 8 * hf;
            const float zv = zb[g * kC + c];
            outh[(size_t)(br * kC + c) * kE + hh * kD + e] = (f16)(acc[s][j] * zv);
        }
    }
}

// ---------------- final projection: out = outh @ Wo^T + bo (f32 out) ----------------
// grid: (M/128, E/128), block 256 (4 waves x 32 M-rows, 2 waves x 64 N-cols).
__global__ void k_final(const f16* __restrict__ a16, const f16* __restrict__ w16,
                        const float* __restrict__ bias, float* __restrict__ out) {
    const int lane = threadIdx.x & 31;
    const int wave = threadIdx.x >> 5;
    const int nl = lane & 15;
    const int hf = lane >> 4;

    const int m0 = blockIdx.x * 128 + (wave & 3) * 32;
    const int n0 = blockIdx.y * 128 + (wave >> 2) * 64;

    const f16* rowA0 = a16 + (size_t)(m0 + nl) * kE;
    const f16* rowA1 = rowA0 + (size_t)16 * kE;
    const f16* rowB[4];
#pragma unroll
    for (int s = 0; s < 4; ++s)
        rowB[s] = w16 + (size_t)(n0 + s * 16 + nl) * kE;

    const v8f zero8 = {0.f,0.f,0.f,0.f,0.f,0.f,0.f,0.f};
    v8f acc0[4] = {zero8, zero8, zero8, zero8};
    v8f acc1[4] = {zero8, zero8, zero8, zero8};

    for (int k0 = 0; k0 < kE; k0 += 32) {
        if (k0 + 64 <= kE) {
            __builtin_prefetch(rowA0 + k0 + 32, 0, 3);
            __builtin_prefetch(rowA1 + k0 + 32, 0, 3);
        }
        v16h a0 = load_frag_a(rowA0, k0, hf);
        v16h a1 = load_frag_a(rowA1, k0, hf);
#pragma unroll
        for (int s = 0; s < 4; ++s) {
            v16h b = load_frag_b(rowB[s], k0, hf);
            acc0[s] = wmma_f16(a0, b, acc0[s]);
            acc1[s] = wmma_f16(a1, b, acc1[s]);
        }
    }

#pragma unroll
    for (int s = 0; s < 4; ++s) {
        const int n = n0 + s * 16 + nl;
        const float bval = bias[n];
#pragma unroll
        for (int j = 0; j < 8; ++j) {
            const int ma = m0 + j + 8 * hf;
            const int mb = ma + 16;
            out[(size_t)ma * kE + n] = acc0[s][j] + bval;
            out[(size_t)mb * kE + n] = acc1[s][j] + bval;
        }
    }
}

// ---------------- host launcher ----------------
extern "C" void kernel_launch(void* const* d_in, const int* in_sizes, int n_in,
                              void* d_out, int out_size, void* d_ws, size_t ws_size,
                              hipStream_t stream) {
    const float* x  = (const float*)d_in[0];
    const float* Wq = (const float*)d_in[1];
    const float* bq = (const float*)d_in[2];
    const float* Wk = (const float*)d_in[3];
    const float* bk = (const float*)d_in[4];
    const float* Wv = (const float*)d_in[5];
    const float* bv = (const float*)d_in[6];
    const float* Wo = (const float*)d_in[7];
    const float* bo = (const float*)d_in[8];
    float* out = (float*)d_out;

    // workspace carve-up (f16 segments first, then f32)
    char* p = (char*)d_ws;
    f16* x16   = (f16*)p; p += (size_t)kMTOT * kE * sizeof(f16);
    f16* q16   = (f16*)p; p += (size_t)kMTOT * kE * sizeof(f16);
    f16* kT16  = (f16*)p; p += (size_t)kG * kD * kC * sizeof(f16);
    f16* vT16  = (f16*)p; p += (size_t)kG * kD * kC * sizeof(f16);
    f16* outh16= (f16*)p; p += (size_t)kMTOT * kE * sizeof(f16);
    f16* ktvT16= (f16*)p; p += (size_t)kG * kD * kD * sizeof(f16);
    f16* wq16  = (f16*)p; p += (size_t)kE * kE * sizeof(f16);
    f16* wk16  = (f16*)p; p += (size_t)kE * kE * sizeof(f16);
    f16* wv16  = (f16*)p; p += (size_t)kE * kE * sizeof(f16);
    f16* wo16  = (f16*)p; p += (size_t)kE * kE * sizeof(f16);
    float* ksum= (float*)p; p += (size_t)kG * kD * sizeof(float);
    float* zb  = (float*)p; p += (size_t)kG * kC * sizeof(float);

    const int nx8 = (kMTOT * kE) / 8;   // 4,194,304 threads (8 elems each)
    const int nw8 = (kE * kE) / 8;      // 32,768

    k_f32_to_f16<<<(nx8 + 255) / 256, 256, 0, stream>>>(x,  x16,  nx8);
    k_f32_to_f16<<<(nw8 + 255) / 256, 256, 0, stream>>>(Wq, wq16, nw8);
    k_f32_to_f16<<<(nw8 + 255) / 256, 256, 0, stream>>>(Wk, wk16, nw8);
    k_f32_to_f16<<<(nw8 + 255) / 256, 256, 0, stream>>>(Wv, wv16, nw8);
    k_f32_to_f16<<<(nw8 + 255) / 256, 256, 0, stream>>>(Wo, wo16, nw8);

    k_qkv<<<dim3(kMTOT / 128, kE / 128, 3), 256, 0, stream>>>(
        x16, wq16, wk16, wv16, bq, bk, bv, q16, kT16, vT16);

    k_ksum<<<(kG * kD) / 256, 256, 0, stream>>>(kT16, ksum);
    k_z<<<(kG * kC) / 256, 256, 0, stream>>>(q16, ksum, zb);

    k_ktv<<<kG, 128, 0, stream>>>(vT16, kT16, ktvT16);

    k_outh<<<dim3(kC / 128, kG), 256, 0, stream>>>(q16, ktvT16, zb, outh16);

    k_final<<<dim3(kMTOT / 128, kE / 128), 256, 0, stream>>>(outh16, wo16, bo, out);
}
